// TPTransformerParallelBlock_35321811043116
// MI455X (gfx1250) — matrix operand order repaired and checked by reading, not compiled, not run on hardware
//
#include <hip/hip_runtime.h>
#include <cstdint>
#include <cstddef>

typedef __bf16 bf16_t;
typedef __attribute__((ext_vector_type(16))) __bf16 bf16x16;
typedef __attribute__((ext_vector_type(8)))  float  floatx8;
typedef __attribute__((ext_vector_type(4)))  int    v4i_t;

union Frag { bf16x16 v; uint4 u[2]; };

// ---- CDNA5 async global->LDS copy (ASYNCcnt) with portable fallback --------
#if defined(__gfx1250__) && __has_builtin(__builtin_amdgcn_global_load_async_to_lds_b128) && __has_builtin(__builtin_amdgcn_s_wait_asynccnt)
#define HAVE_ASYNC_LDS 1
#else
#define HAVE_ASYNC_LDS 0
#endif

__device__ __forceinline__ void gld_async_b128(void* lds_dst, const void* gsrc) {
#if HAVE_ASYNC_LDS
  __builtin_amdgcn_global_load_async_to_lds_b128(
      (__attribute__((address_space(1))) v4i_t*)gsrc,
      (__attribute__((address_space(3))) v4i_t*)lds_dst, 0, 0);
#else
  *(uint4*)lds_dst = *(const uint4*)gsrc;
#endif
}

__device__ __forceinline__ void async_wait_all() {
#if HAVE_ASYNC_LDS
  __builtin_amdgcn_s_wait_asynccnt(0);
#endif
}

__device__ __forceinline__ floatx8 zero8() {
  floatx8 z = {0.f,0.f,0.f,0.f,0.f,0.f,0.f,0.f};
  return z;
}

__device__ __forceinline__ float gelu_tanh(float x) {
  const float c = 0.7978845608028654f;   // sqrt(2/pi)
  return 0.5f * x * (1.0f + tanhf(c * (x + 0.044715f * x * x * x)));
}

// ---------------------------------------------------------------------------
// f32 -> bf16 elementwise convert (for weights)
// ---------------------------------------------------------------------------
__global__ __launch_bounds__(256) void f32_to_bf16_kernel(
    const float* __restrict__ in, bf16_t* __restrict__ out, size_t n) {
  size_t i = (size_t)blockIdx.x * 256 + threadIdx.x;
  size_t stride = (size_t)gridDim.x * 256;
  for (; i < n; i += stride) out[i] = (bf16_t)in[i];
}

// ---------------------------------------------------------------------------
// RMSNorm row kernel: one block per row of D=1024, writes bf16
// ---------------------------------------------------------------------------
__global__ __launch_bounds__(256) void rmsnorm_bf16_kernel(
    const float* __restrict__ x, const float* __restrict__ scale,
    bf16_t* __restrict__ out, int D) {
  const int row = blockIdx.x;
  const float* xr = x + (size_t)row * D;
  float ss = 0.f;
  for (int i = threadIdx.x; i < D; i += 256) { float v = xr[i]; ss += v * v; }
  __shared__ float red[8];
  for (int o = 16; o > 0; o >>= 1) ss += __shfl_down(ss, o, 32);
  if ((threadIdx.x & 31) == 0) red[threadIdx.x >> 5] = ss;
  __syncthreads();
  if (threadIdx.x < 8) {
    float v = red[threadIdx.x];
    for (int o = 4; o > 0; o >>= 1) v += __shfl_down(v, o, 8);
    if (threadIdx.x == 0) red[0] = v;
  }
  __syncthreads();
  const float inv = rsqrtf(red[0] / (float)D + 1e-6f);
  for (int i = threadIdx.x; i < D; i += 256)
    out[(size_t)row * D + i] = (bf16_t)(xr[i] * inv * scale[i]);
}

// ---------------------------------------------------------------------------
// out = x + b_mlp_out[d] + b_attn_out[d]   (residual + biases, f32)
// ---------------------------------------------------------------------------
__global__ __launch_bounds__(256) void init_out_kernel(
    const float* __restrict__ x, const float* __restrict__ bm,
    const float* __restrict__ ba, float* __restrict__ out, size_t total) {
  size_t i = (size_t)blockIdx.x * 256 + threadIdx.x;
  size_t stride = (size_t)gridDim.x * 256;
  for (; i < total; i += stride) {
    int d = (int)(i & 1023);           // D = 1024
    out[i] = x[i] + bm[d] + ba[d];
  }
}

// ---------------------------------------------------------------------------
// WMMA GEMM: C[M,N] = A[M,K] (bf16, row-major) * B[K,N] (bf16, row-major)
// Block tile 128x128, BK=32, 256 threads = 8 waves, each wave 32x64.
// MODE 0: write bf16   MODE 1: gelu -> bf16   MODE 2: f32 accumulate-add
// ---------------------------------------------------------------------------
template <int MODE>
__global__ __launch_bounds__(256) void wmma_gemm_kernel(
    const bf16_t* __restrict__ A, const bf16_t* __restrict__ B,
    void* __restrict__ Cout, int M, int N, int K) {
  const int tid  = threadIdx.x;
  const int lane = tid & 31;
  const int wid  = tid >> 5;
  const int wm   = wid & 3;            // 4 waves along M, 32 rows each
  const int wn   = wid >> 2;           // 2 waves along N, 64 cols each
  const int bm   = blockIdx.y * 128;
  const int bn   = blockIdx.x * 128;

  __shared__ __align__(16) bf16_t As[128 * 40];   // [m][k], padded stride 40
  __shared__ __align__(16) bf16_t Bs[128 * 40];   // [n][k] (transposed), stride 40

  floatx8 acc[2][4];
#pragma unroll
  for (int i = 0; i < 2; ++i)
#pragma unroll
    for (int j = 0; j < 4; ++j) acc[i][j] = zero8();

  const int arow = tid >> 1;           // 0..127
  const int acol = (tid & 1) * 16;     // 0 or 16
  const int brow = tid >> 3;           // 0..31 (k)
  const int bcol = (tid & 7) * 16;     // 0..112 (n)
  const int r  = lane & 15;
  const int hf = lane >> 4;

  for (int k0 = 0; k0 < K; k0 += 32) {
    // ---- stage A tile (row-major) via async global->LDS ----
    const bf16_t* ag = A + (size_t)(bm + arow) * K + k0 + acol;
    gld_async_b128(As + arow * 40 + acol,     ag);
    gld_async_b128(As + arow * 40 + acol + 8, ag + 8);

    // ---- stage B tile transposed to [n][k] (16-bit scatter) ----
    const bf16_t* bg = B + (size_t)(k0 + brow) * N + bn + bcol;
    uint4 bv0 = *(const uint4*)bg;
    uint4 bv1 = *(const uint4*)(bg + 8);
    const bf16_t* bvp0 = (const bf16_t*)&bv0;
    const bf16_t* bvp1 = (const bf16_t*)&bv1;
#pragma unroll
    for (int i = 0; i < 8; ++i) Bs[(bcol + i) * 40 + brow]     = bvp0[i];
#pragma unroll
    for (int i = 0; i < 8; ++i) Bs[(bcol + 8 + i) * 40 + brow] = bvp1[i];

    if (k0 + 32 < K) {
      __builtin_prefetch(A + (size_t)(bm + arow) * K + k0 + 32 + acol, 0, 1);
      __builtin_prefetch(B + (size_t)(k0 + 32 + brow) * N + bn + bcol, 0, 1);
    }
    async_wait_all();
    __syncthreads();

    // ---- fragments + WMMA (8 per wave per K-step) ----
    Frag af[2], bfr[4];
#pragma unroll
    for (int i = 0; i < 2; ++i) {
      const bf16_t* ap = As + (wm * 32 + i * 16 + r) * 40;
      af[i].u[0] = *(const uint4*)(ap + hf * 8);
      af[i].u[1] = *(const uint4*)(ap + 16 + hf * 8);
    }
#pragma unroll
    for (int j = 0; j < 4; ++j) {
      const bf16_t* bp = Bs + (wn * 64 + j * 16 + r) * 40;
      bfr[j].u[0] = *(const uint4*)(bp + hf * 8);
      bfr[j].u[1] = *(const uint4*)(bp + 16 + hf * 8);
    }
#pragma unroll
    for (int i = 0; i < 2; ++i)
#pragma unroll
      for (int j = 0; j < 4; ++j)
        acc[i][j] = __builtin_amdgcn_wmma_f32_16x16x32_bf16(
            false, af[i].v, false, bfr[j].v, (short)0, acc[i][j], false, false);
    __syncthreads();
  }

  // ---- epilogue (C layout: VGPR e, lanes0-15: M=e; lanes16-31: M=e+8) ----
#pragma unroll
  for (int i = 0; i < 2; ++i)
#pragma unroll
    for (int j = 0; j < 4; ++j) {
      const int gm = bm + wm * 32 + i * 16;
      const int gn = bn + wn * 64 + j * 16 + r;
#pragma unroll
      for (int e = 0; e < 8; ++e) {
        const int row = gm + e + hf * 8;
        float v = acc[i][j][e];
        if (MODE == 1) v = gelu_tanh(v);
        if (MODE == 2) {
          float* o = (float*)Cout;
          o[(size_t)row * N + gn] += v;
        } else {
          bf16_t* o = (bf16_t*)Cout;
          o[(size_t)row * N + gn] = (bf16_t)v;
        }
      }
    }
}

// ---------------------------------------------------------------------------
// Flash attention: grid (S/64, B*H), 128 threads (4 waves), wave = 16 q-rows.
// Q,K,V,O are bf16 [B,S,H=16,HD=64]; softmax scale = 1/8.
// ---------------------------------------------------------------------------
__global__ __launch_bounds__(128) void attention_kernel(
    const bf16_t* __restrict__ Q, const bf16_t* __restrict__ Kb,
    const bf16_t* __restrict__ Vb, bf16_t* __restrict__ O, int S) {
  const int qt = blockIdx.x;
  const int bh = blockIdx.y;
  const int b  = bh >> 4;
  const int h  = bh & 15;
  const int q0 = qt * 64;
  const int tid  = threadIdx.x;
  const int lane = tid & 31;
  const int w    = tid >> 5;           // wave 0..3

  __shared__ __align__(16) bf16_t Qs[64 * 72];   // [q][d]
  __shared__ __align__(16) bf16_t Ks[64 * 72];   // [kv][d] (== B-frag layout for QK^T)
  __shared__ __align__(16) bf16_t Vt[64 * 72];   // [d][kv] (B-frag layout for PV)
  __shared__ __align__(16) float  Ss[4][16 * 72];
  __shared__ __align__(16) bf16_t Ps[4][16 * 72];
  __shared__ float rowf[4][16];

  // load Q tile (row stride in memory: H*HD = 1024 bf16) via async copies
  {
    const int rq = tid >> 1, part = tid & 1;
    const bf16_t* src = Q + ((size_t)(b * S + q0 + rq) * 16 + h) * 64 + part * 32;
    bf16_t* dst = Qs + rq * 72 + part * 32;
#pragma unroll
    for (int c = 0; c < 4; ++c) gld_async_b128(dst + c * 8, src + c * 8);
  }

  const int rr = lane & 15;
  const int hf = lane >> 4;
  float m_run = -1e30f, l_run = 0.f;
  floatx8 o[4];
#pragma unroll
  for (int t = 0; t < 4; ++t) o[t] = zero8();

  for (int kv0 = 0; kv0 < S; kv0 += 64) {
    async_wait_all();
    __syncthreads();
    // ---- load K tile row-major (async) ----
    {
      const int rk = tid >> 1, part = tid & 1;
      const bf16_t* src = Kb + ((size_t)(b * S + kv0 + rk) * 16 + h) * 64 + part * 32;
      bf16_t* dst = Ks + rk * 72 + part * 32;
#pragma unroll
      for (int c = 0; c < 4; ++c) gld_async_b128(dst + c * 8, src + c * 8);
    }
    // ---- load V tile transposed to [d][kv] (scatter) ----
    {
      const int kvr = tid >> 1, dbase = (tid & 1) * 32;
      const bf16_t* src = Vb + ((size_t)(b * S + kv0 + kvr) * 16 + h) * 64 + dbase;
      uint4 vv[4];
      vv[0] = ((const uint4*)src)[0]; vv[1] = ((const uint4*)src)[1];
      vv[2] = ((const uint4*)src)[2]; vv[3] = ((const uint4*)src)[3];
      const bf16_t* e = (const bf16_t*)vv;
#pragma unroll
      for (int i = 0; i < 32; ++i) Vt[(dbase + i) * 72 + kvr] = e[i];
    }
    async_wait_all();
    __syncthreads();

    // ---- scores = Q @ K^T (16 q-rows x 64 kv-cols per wave) ----
#pragma unroll
    for (int ns = 0; ns < 4; ++ns) {
      floatx8 s = zero8();
#pragma unroll
      for (int ks = 0; ks < 2; ++ks) {
        Frag af, bfr;
        const bf16_t* ap = Qs + (w * 16 + rr) * 72 + ks * 32;
        af.u[0] = *(const uint4*)(ap + hf * 8);
        af.u[1] = *(const uint4*)(ap + 16 + hf * 8);
        const bf16_t* bp = Ks + (ns * 16 + rr) * 72 + ks * 32;
        bfr.u[0] = *(const uint4*)(bp + hf * 8);
        bfr.u[1] = *(const uint4*)(bp + 16 + hf * 8);
        s = __builtin_amdgcn_wmma_f32_16x16x32_bf16(
            false, af.v, false, bfr.v, (short)0, s, false, false);
      }
#pragma unroll
      for (int e = 0; e < 8; ++e)
        Ss[w][(e + hf * 8) * 72 + ns * 16 + rr] = s[e];
    }
    __syncthreads();

    // ---- online softmax: lanes 0..15 own one q-row each ----
    if (lane < 16) {
      const float* sr = &Ss[w][lane * 72];
      float mx = -1e30f;
#pragma unroll
      for (int j = 0; j < 64; ++j) mx = fmaxf(mx, sr[j]);
      mx *= 0.125f;
      const float mnew = fmaxf(m_run, mx);
      const float corr = __expf(m_run - mnew);
      float lsum = 0.f;
#pragma unroll
      for (int j = 0; j < 64; ++j) {
        float p = __expf(sr[j] * 0.125f - mnew);
        lsum += p;
        Ps[w][lane * 72 + j] = (bf16_t)p;
      }
      l_run = l_run * corr + lsum;
      m_run = mnew;
      rowf[w][lane] = corr;
    }
    __syncthreads();

    // ---- rescale running O by correction factors ----
    float cf[8];
#pragma unroll
    for (int e = 0; e < 8; ++e) cf[e] = rowf[w][hf * 8 + e];
#pragma unroll
    for (int t = 0; t < 4; ++t)
#pragma unroll
      for (int e = 0; e < 8; ++e) o[t][e] *= cf[e];

    // ---- O += P @ V ----
#pragma unroll
    for (int ds = 0; ds < 4; ++ds) {
#pragma unroll
      for (int ks = 0; ks < 2; ++ks) {
        Frag af, bfr;
        const bf16_t* ap = &Ps[w][rr * 72 + ks * 32];
        af.u[0] = *(const uint4*)(ap + hf * 8);
        af.u[1] = *(const uint4*)(ap + 16 + hf * 8);
        const bf16_t* bp = Vt + (ds * 16 + rr) * 72 + ks * 32;
        bfr.u[0] = *(const uint4*)(bp + hf * 8);
        bfr.u[1] = *(const uint4*)(bp + 16 + hf * 8);
        o[ds] = __builtin_amdgcn_wmma_f32_16x16x32_bf16(
            false, af.v, false, bfr.v, (short)0, o[ds], false, false);
      }
    }
  }

  // ---- finalize: O /= l, write bf16 ----
  __syncthreads();
  if (lane < 16) rowf[w][lane] = 1.0f / l_run;
  __syncthreads();
  float lf[8];
#pragma unroll
  for (int e = 0; e < 8; ++e) lf[e] = rowf[w][hf * 8 + e];
#pragma unroll
  for (int ds = 0; ds < 4; ++ds)
#pragma unroll
    for (int e = 0; e < 8; ++e) {
      const int row = q0 + w * 16 + e + hf * 8;
      const int col = ds * 16 + rr;
      O[((size_t)(b * S + row) * 16 + h) * 64 + col] = (bf16_t)(o[ds][e] * lf[e]);
    }
}

// ---------------------------------------------------------------------------
// host-side launch
// ---------------------------------------------------------------------------
extern "C" void kernel_launch(void* const* d_in, const int* in_sizes, int n_in,
                              void* d_out, int out_size, void* d_ws, size_t ws_size,
                              hipStream_t stream) {
  const int   S = 2048, D = 1024, MLPD = 4096;
  const size_t M = 8192;                         // B * S

  const float* x     = (const float*)d_in[0];
  const float* scale = (const float*)d_in[1];
  const float* w_mlp_in  = (const float*)d_in[2];
  const float* wq        = (const float*)d_in[3];
  const float* wk        = (const float*)d_in[4];
  const float* wv        = (const float*)d_in[5];
  const float* w_mlp_out = (const float*)d_in[6];
  const float* b_mlp_out = (const float*)d_in[7];
  const float* w_attn_out= (const float*)d_in[8];
  const float* b_attn_out= (const float*)d_in[9];
  float* out = (float*)d_out;

  // workspace carve-up (bytes)
  char* ws = (char*)d_ws;
  size_t off = 0;
  auto carve = [&](size_t bytes) { char* p = ws + off; off += (bytes + 255) & ~(size_t)255; return p; };
  bf16_t* xn   = (bf16_t*)carve(M * D * 2);          // 16 MB
  bf16_t* hb   = (bf16_t*)carve(M * MLPD * 2);       // 64 MB
  bf16_t* qb   = (bf16_t*)carve(M * D * 2);
  bf16_t* kb   = (bf16_t*)carve(M * D * 2);
  bf16_t* vb   = (bf16_t*)carve(M * D * 2);
  bf16_t* avb  = (bf16_t*)carve(M * D * 2);
  bf16_t* wmi  = (bf16_t*)carve((size_t)D * MLPD * 2);
  bf16_t* wqb  = (bf16_t*)carve((size_t)D * D * 2);
  bf16_t* wkb  = (bf16_t*)carve((size_t)D * D * 2);
  bf16_t* wvb  = (bf16_t*)carve((size_t)D * D * 2);
  bf16_t* wmo  = (bf16_t*)carve((size_t)MLPD * D * 2);
  bf16_t* wao  = (bf16_t*)carve((size_t)D * D * 2);

  // 1. weight conversions to bf16
  f32_to_bf16_kernel<<<2048, 256, 0, stream>>>(w_mlp_in,  wmi, (size_t)D * MLPD);
  f32_to_bf16_kernel<<<1024, 256, 0, stream>>>(wq,        wqb, (size_t)D * D);
  f32_to_bf16_kernel<<<1024, 256, 0, stream>>>(wk,        wkb, (size_t)D * D);
  f32_to_bf16_kernel<<<1024, 256, 0, stream>>>(wv,        wvb, (size_t)D * D);
  f32_to_bf16_kernel<<<2048, 256, 0, stream>>>(w_mlp_out, wmo, (size_t)MLPD * D);
  f32_to_bf16_kernel<<<1024, 256, 0, stream>>>(w_attn_out,wao, (size_t)D * D);

  // 2. RMSNorm -> bf16
  rmsnorm_bf16_kernel<<<(int)M, 256, 0, stream>>>(x, scale, xn, D);

  // 3. h = gelu(xn @ w_mlp_in)   [8192 x 4096]
  wmma_gemm_kernel<1><<<dim3(MLPD / 128, (int)(M / 128)), 256, 0, stream>>>(
      xn, wmi, hb, (int)M, MLPD, D);

  // 4. q/k/v = xn @ w{q,k,v}     [8192 x 1024]
  wmma_gemm_kernel<0><<<dim3(D / 128, (int)(M / 128)), 256, 0, stream>>>(xn, wqb, qb, (int)M, D, D);
  wmma_gemm_kernel<0><<<dim3(D / 128, (int)(M / 128)), 256, 0, stream>>>(xn, wkb, kb, (int)M, D, D);
  wmma_gemm_kernel<0><<<dim3(D / 128, (int)(M / 128)), 256, 0, stream>>>(xn, wvb, vb, (int)M, D, D);

  // 5. flash attention -> attn_v
  attention_kernel<<<dim3(S / 64, 64), 128, 0, stream>>>(qb, kb, vb, avb, S);

  // 6. out = residual + biases
  init_out_kernel<<<4096, 256, 0, stream>>>(x, b_mlp_out, b_attn_out, out, M * D);

  // 7. out += h @ w_mlp_out      (K = 4096)
  wmma_gemm_kernel<2><<<dim3(D / 128, (int)(M / 128)), 256, 0, stream>>>(hb, wmo, out, (int)M, D, MLPD);

  // 8. out += attn_v @ w_attn_out (K = 1024)
  wmma_gemm_kernel<2><<<dim3(D / 128, (int)(M / 128)), 256, 0, stream>>>(avb, wao, out, (int)M, D, D);
}
